// G3Median_GCNConv_62989990363518
// MI455X (gfx1250) — matrix operand order repaired and checked by reading, not compiled, not run on hardware
//
#include <hip/hip_runtime.h>

#define NNODES 100000
#define NEDGES 1000000
#define HID    75
#define OUTC   64
#define KPAD   96    // K padded to 3 x 32 for wmma f16 16x16x32
#define CPAD   80    // fp32 intermediate row stride (5 x 16 col tiles)
#define NT     5     // column tiles (heads pack tile 4 as zeros)
#define BN_EPS 1e-5f

typedef __attribute__((ext_vector_type(16))) _Float16 v16h;
typedef __attribute__((ext_vector_type(8)))  _Float16 v8h;
typedef __attribute__((ext_vector_type(8)))  float    v8f;

static __device__ __forceinline__ v16h cat8(v8h lo, v8h hi) {
  return __builtin_shufflevector(lo, hi, 0,1,2,3,4,5,6,7,8,9,10,11,12,13,14,15);
}

// ---------------------------------------------------------------- degree prep
__global__ void k_fill_one(float* __restrict__ d, int n) {
  int i = blockIdx.x * blockDim.x + threadIdx.x;
  if (i < n) d[i] = 1.0f;                       // self-loop contribution to deg
}

__global__ void k_deg_count(const int* __restrict__ dst, float* __restrict__ deg, int e) {
  int i = blockIdx.x * blockDim.x + threadIdx.x;
  if (i < e) atomicAdd(&deg[dst[i]], 1.0f);
}

__global__ void k_rsqrt_inplace(float* __restrict__ d, int n) {
  int i = blockIdx.x * blockDim.x + threadIdx.x;
  if (i < n) d[i] = rsqrtf(d[i]);               // deg >= 1 always (self-loop)
}

__global__ void k_edge_norm(const int* __restrict__ src, const int* __restrict__ dst,
                            const float* __restrict__ dinv, float* __restrict__ norm, int e) {
  int i = blockIdx.x * blockDim.x + threadIdx.x;
  if (i < e) norm[i] = dinv[src[i]] * dinv[dst[i]];
}

// ------------------------------------------------------------ embedding gather
__global__ void k_embed(const int* __restrict__ x, const float* __restrict__ emb,
                        _Float16* __restrict__ hf, unsigned total) {
  unsigned idx = blockIdx.x * blockDim.x + threadIdx.x;
  if (idx >= total) return;
  unsigned i = idx / KPAD, k = idx - i * KPAD;
  float v = (k < HID) ? emb[(size_t)x[i] * HID + k] : 0.0f;
  hf[idx] = (_Float16)v;
}

// ----------------------------------------------- pack weight into WMMA B-frags
// bfrag layout: [ntile][kchunk][lane 0..31][elem 0..15]
// B 32x16 f16 layout: col = lane&15; lanes<16 hold K {0..7,16..23}, lanes>=16 hold K {8..15,24..31},
// element e of the lane's v16h holds K = kc*32 + (lane<16?0:8) + (e<8 ? e : e+8).
__global__ void k_pack_bfrag(const float* __restrict__ W, _Float16* __restrict__ bfrag,
                             int ncols, int total) {
  int idx = blockIdx.x * blockDim.x + threadIdx.x;
  if (idx >= total) return;
  int e    = idx & 15;
  int lane = (idx >> 4) & 31;
  int kc   = (idx >> 9) % 3;
  int nt   = (idx >> 9) / 3;
  int n = nt * 16 + (lane & 15);
  int k = kc * 32 + ((lane & 16) ? 8 : 0) + (e < 8 ? e : e + 8);
  float v = (k < HID && n < ncols) ? W[k * ncols + n] : 0.0f;
  bfrag[idx] = (_Float16)v;
}

// ------------------------------------------------------------------ WMMA GEMM
// Persistent waves: B (all NT x 3 fragments) held in registers for the whole
// kernel; A fragments double-buffered so global loads overlap the WMMAs.
__global__ void __launch_bounds__(128)
k_wmma_gemm(const _Float16* __restrict__ hf, const _Float16* __restrict__ bfrag,
            float* __restrict__ out, int nrows, int ncols, int ostride) {
  const int lane = threadIdx.x & 31;
  const int wavesTotal = gridDim.x * (blockDim.x >> 5);
  const int wave = blockIdx.x * (blockDim.x >> 5) + (threadIdx.x >> 5);
  const int nstripes = nrows >> 4;              // nrows is a multiple of 16

  int m = wave;
  if (m >= nstripes) return;                    // wave-uniform: EXEC stays all-1s

  // B fragments: loaded once, reused for every stripe this wave processes.
  v16h B[NT * 3];
#pragma unroll
  for (int t = 0; t < NT * 3; ++t)
    B[t] = *(const v16h*)(bfrag + (size_t)t * 512 + lane * 16);

  // A 16x32 f16 layout: lanes<16 -> row=lane, K {0..7,16..23}+kb;
  //                     lanes>=16 -> row=lane-16, K {8..15,24..31}+kb.
  const int lrow = lane & 15;
  const int koff = (lane & 16) ? 8 : 0;
  const int rsub = (lane & 16) ? 8 : 0;

  const _Float16* rowp = hf + (size_t)(m * 16 + lrow) * KPAD + koff;
  v16h a0 = cat8(*(const v8h*)(rowp +  0), *(const v8h*)(rowp + 16));
  v16h a1 = cat8(*(const v8h*)(rowp + 32), *(const v8h*)(rowp + 48));
  v16h a2 = cat8(*(const v8h*)(rowp + 64), *(const v8h*)(rowp + 80));

  while (true) {
    const int mn = m + wavesTotal;
    const bool more = (mn < nstripes);          // wave-uniform
    v16h n0 = {}, n1 = {}, n2 = {};
    if (more) {                                 // prefetch next stripe's A
      const _Float16* rp = hf + (size_t)(mn * 16 + lrow) * KPAD + koff;
      n0 = cat8(*(const v8h*)(rp +  0), *(const v8h*)(rp + 16));
      n1 = cat8(*(const v8h*)(rp + 32), *(const v8h*)(rp + 48));
      n2 = cat8(*(const v8h*)(rp + 64), *(const v8h*)(rp + 80));
    }
#pragma unroll
    for (int nt = 0; nt < NT; ++nt) {
      v8f c = {};
      c = __builtin_amdgcn_wmma_f32_16x16x32_f16(false, a0, false, B[nt*3+0], (short)0, c, false, false);
      c = __builtin_amdgcn_wmma_f32_16x16x32_f16(false, a1, false, B[nt*3+1], (short)0, c, false, false);
      c = __builtin_amdgcn_wmma_f32_16x16x32_f16(false, a2, false, B[nt*3+2], (short)0, c, false, false);
      // C/D f32 16x16: lane holds col (lane&15); VGPR r = row r / row 8+r.
      int n = nt * 16 + lrow;
      if (n < ncols) {
        float* op = out + (size_t)(m * 16 + rsub) * ostride + n;
#pragma unroll
        for (int r = 0; r < 8; ++r) op[(size_t)r * ostride] = c[r];
      }
    }
    if (!more) break;
    a0 = n0; a1 = n1; a2 = n2; m = mn;
  }
}

// ---------------------------------------------- self-loop + bias (non-atomic)
__global__ void k_init_agg(const float* __restrict__ hw, const float* __restrict__ dinv,
                           const float* __restrict__ b, float* __restrict__ agg,
                           int ncols, int istride, int ostride, unsigned total) {
  unsigned idx = blockIdx.x * blockDim.x + threadIdx.x;
  if (idx >= total) return;
  unsigned i = idx / ncols, f = idx - i * ncols;
  float d = dinv[i];
  agg[(size_t)i * ostride + f] = hw[(size_t)i * istride + f] * d * d + b[f];
}

// ----------------------------------------------------- edge scatter (atomics)
__global__ void k_scatter(const int* __restrict__ src, const int* __restrict__ dst,
                          const float* __restrict__ norm, const float* __restrict__ hw,
                          float* __restrict__ agg, int ncols, int istride, int ostride,
                          unsigned total) {
  unsigned idx = blockIdx.x * blockDim.x + threadIdx.x;
  if (idx >= total) return;
  unsigned e = idx / ncols, f = idx - e * ncols;
  float v = hw[(size_t)src[e] * istride + f] * norm[e];
  atomicAdd(&agg[(size_t)dst[e] * ostride + f], v);
}

// -------------------------------------------------------------------- BN pass
__global__ void k_zero(float* __restrict__ p, int n) {
  int i = blockIdx.x * blockDim.x + threadIdx.x;
  if (i < n) p[i] = 0.0f;
}

// Grid chosen so gridDim*blockDim is a multiple of HID: each thread owns a
// fixed feature column and accumulates in registers; 2 LDS atomics per thread.
__global__ void k_bn_stats(const float* __restrict__ agg, float* __restrict__ stats) {
  __shared__ float ssum[HID], ssq[HID];
  if (threadIdx.x < HID) { ssum[threadIdx.x] = 0.f; ssq[threadIdx.x] = 0.f; }
  __syncthreads();
  unsigned tid  = blockIdx.x * blockDim.x + threadIdx.x;
  unsigned f    = tid % HID;
  unsigned i    = tid / HID;
  unsigned istp = (gridDim.x * blockDim.x) / HID;   // exact (multiple of HID)
  float s = 0.f, q = 0.f;
  for (; i < NNODES; i += istp) {
    float v = agg[(size_t)i * CPAD + f];
    s += v; q += v * v;
  }
  atomicAdd(&ssum[f], s);
  atomicAdd(&ssq[f], q);
  __syncthreads();
  if (threadIdx.x < HID) {
    atomicAdd(&stats[threadIdx.x], ssum[threadIdx.x]);
    atomicAdd(&stats[CPAD + threadIdx.x], ssq[threadIdx.x]);
  }
}

__global__ void k_bn_relu_tof16(const float* __restrict__ agg, const float* __restrict__ stats,
                                const float* __restrict__ gamma, const float* __restrict__ beta,
                                _Float16* __restrict__ hf, int stride, unsigned total) {
  unsigned idx = blockIdx.x * blockDim.x + threadIdx.x;
  if (idx >= total) return;
  unsigned i = idx / KPAD, k = idx - i * KPAD;
  float o = 0.0f;
  if (k < HID) {
    const float invn = 1.0f / (float)NNODES;
    float mu  = stats[k] * invn;
    float var = stats[CPAD + k] * invn - mu * mu;
    float v = (agg[(size_t)i * stride + k] - mu) * rsqrtf(var + BN_EPS) * gamma[k] + beta[k];
    o = fmaxf(v, 0.0f);
  }
  hf[idx] = (_Float16)o;
}

// ============================================================================
extern "C" void kernel_launch(void* const* d_in, const int* in_sizes, int n_in,
                              void* d_out, int out_size, void* d_ws, size_t ws_size,
                              hipStream_t stream) {
  const int*   x        = (const int*)d_in[0];
  const int*   esrc     = (const int*)d_in[1];
  const int*   edst     = esrc + NEDGES;
  const float* emb      = (const float*)d_in[2];
  const float* Ws       = (const float*)d_in[3];
  const float* bs       = (const float*)d_in[4];
  const float* gammas   = (const float*)d_in[5];
  const float* betas    = (const float*)d_in[6];
  const float* W_mu     = (const float*)d_in[7];
  const float* b_mu     = (const float*)d_in[8];
  const float* W_logstd = (const float*)d_in[9];
  const float* b_logstd = (const float*)d_in[10];
  float* out_mu     = (float*)d_out;
  float* out_logstd = out_mu + (size_t)NNODES * OUTC;

  // workspace carve-out (256B aligned)
  char* ws = (char*)d_ws;
  size_t off = 0;
  auto carve = [&](size_t bytes) { void* p = ws + off; off = (off + bytes + 255) & ~(size_t)255; return p; };
  float*    dinv  = (float*)   carve(sizeof(float) * NNODES);
  float*    enorm = (float*)   carve(sizeof(float) * NEDGES);
  _Float16* hf    = (_Float16*)carve(sizeof(_Float16) * (size_t)NNODES * KPAD);
  float*    hw    = (float*)   carve(sizeof(float) * (size_t)NNODES * CPAD);
  float*    agg   = (float*)   carve(sizeof(float) * (size_t)NNODES * CPAD);
  float*    stats = (float*)   carve(sizeof(float) * 2 * CPAD);
  _Float16* bfrag = (_Float16*)carve(sizeof(_Float16) * NT * 3 * 512);
  (void)ws_size; (void)in_sizes; (void)n_in; (void)out_size;

  const int T = 256;
  auto blk = [](unsigned n, int t) { return (n + t - 1) / t; };

  // --- degree / norm precompute (layer-invariant) ---
  k_fill_one     <<<blk(NNODES, T), T, 0, stream>>>(dinv, NNODES);
  k_deg_count    <<<blk(NEDGES, T), T, 0, stream>>>(edst, dinv, NEDGES);
  k_rsqrt_inplace<<<blk(NNODES, T), T, 0, stream>>>(dinv, NNODES);
  k_edge_norm    <<<blk(NEDGES, T), T, 0, stream>>>(esrc, edst, dinv, enorm, NEDGES);

  // --- embedding -> fp16 padded activations ---
  const unsigned hfTot = (unsigned)NNODES * KPAD;
  k_embed<<<blk(hfTot, T), T, 0, stream>>>(x, emb, hf, hfTot);

  const int gemmBlocks = 256;                    // 1024 persistent waves
  const int packTot    = NT * 3 * 512;
  const unsigned aggTot = (unsigned)NNODES * HID;
  const unsigned sctTot = (unsigned)NEDGES * HID;

  // --- 4 hidden GCN layers ---
  for (int l = 0; l < 4; ++l) {
    k_pack_bfrag<<<blk(packTot, T), T, 0, stream>>>(Ws + (size_t)l * HID * HID, bfrag, HID, packTot);
    k_wmma_gemm<<<gemmBlocks, 128, 0, stream>>>(hf, bfrag, hw, NNODES, HID, CPAD);
    k_init_agg<<<blk(aggTot, T), T, 0, stream>>>(hw, dinv, bs + (size_t)l * HID, agg, HID, CPAD, CPAD, aggTot);
    k_scatter <<<blk(sctTot, T), T, 0, stream>>>(esrc, edst, enorm, hw, agg, HID, CPAD, CPAD, sctTot);
    k_zero    <<<1, 2 * CPAD, 0, stream>>>(stats, 2 * CPAD);
    k_bn_stats<<<900, T, 0, stream>>>(agg, stats);   // 900*256 = 75*3072
    k_bn_relu_tof16<<<blk(hfTot, T), T, 0, stream>>>(agg, stats, gammas + (size_t)l * HID,
                                                     betas + (size_t)l * HID, hf, CPAD, hfTot);
  }

  // --- heads: write straight into d_out ---
  const unsigned aggTotO = (unsigned)NNODES * OUTC;
  const unsigned sctTotO = (unsigned)NEDGES * OUTC;

  k_pack_bfrag<<<blk(packTot, T), T, 0, stream>>>(W_mu, bfrag, OUTC, packTot);
  k_wmma_gemm<<<gemmBlocks, 128, 0, stream>>>(hf, bfrag, hw, NNODES, OUTC, CPAD);
  k_init_agg<<<blk(aggTotO, T), T, 0, stream>>>(hw, dinv, b_mu, out_mu, OUTC, CPAD, OUTC, aggTotO);
  k_scatter <<<blk(sctTotO, T), T, 0, stream>>>(esrc, edst, enorm, hw, out_mu, OUTC, CPAD, OUTC, sctTotO);

  k_pack_bfrag<<<blk(packTot, T), T, 0, stream>>>(W_logstd, bfrag, OUTC, packTot);
  k_wmma_gemm<<<gemmBlocks, 128, 0, stream>>>(hf, bfrag, hw, NNODES, OUTC, CPAD);
  k_init_agg<<<blk(aggTotO, T), T, 0, stream>>>(hw, dinv, b_logstd, out_logstd, OUTC, CPAD, OUTC, aggTotO);
  k_scatter <<<blk(sctTotO, T), T, 0, stream>>>(esrc, edst, enorm, hw, out_logstd, OUTC, CPAD, OUTC, sctTotO);
}
